// GCNII_58480274703250
// MI455X (gfx1250) — matrix operand order repaired and checked by reading, not compiled, not run on hardware
//
#include <hip/hip_runtime.h>
#include <hip/hip_bf16.h>
#include <cmath>

// ---------------------------------------------------------------------------
// GCNII forward for MI455X (gfx1250, wave32).
//  - Dense GEMMs: v_wmma_f32_16x16x32_bf16, fully unrolled K loop,
//    A tile staged in LDS in fragment order (bf16), B pre-packed in
//    fragment order so each lane issues one 32B vector load per k-step.
//  - Sparse prop: one wave per edge, float4 per lane, global_atomic_add_f32.
//  - Working set (~150MB) is L2-resident (192MB global L2).
// ---------------------------------------------------------------------------

typedef __attribute__((ext_vector_type(16))) __bf16 v16bf;
typedef __attribute__((ext_vector_type(8)))  float  v8f;

#define HDIM    128
#define ALPHA_C 0.1f
#define THETA_C 0.5f
#define EPS_C   1e-5f

// -------------------------------- utility ----------------------------------

__global__ void fill_f32(float* __restrict__ p, float v, size_t n) {
  size_t i = (size_t)blockIdx.x * blockDim.x + threadIdx.x;
  if (i < n) p[i] = v;
}

// ------------------------------ degree / norm ------------------------------

__global__ void deg_accum(const int* __restrict__ cols, float* __restrict__ deg,
                          int ne) {
  int i = blockIdx.x * blockDim.x + threadIdx.x;
  if (i < ne) atomicAdd(&deg[cols[i]], 1.0f);
}

__global__ void deg_to_dinv(float* __restrict__ deg, int n) {
  int i = blockIdx.x * blockDim.x + threadIdx.x;
  if (i < n) { float d = deg[i]; deg[i] = (d > 0.0f) ? rsqrtf(d) : 0.0f; }
}

// ------------------------------- propagation -------------------------------

// out[node][c] = dinv[node]^2 * h[node][c] (+ bias[c])   (self-loop + init)
__global__ void prop_init(const float* __restrict__ dinv,
                          const float* __restrict__ h,
                          const float* __restrict__ bias,
                          float* __restrict__ out, size_t total) {
  size_t i = (size_t)blockIdx.x * blockDim.x + threadIdx.x;
  if (i < total) {
    int node = (int)(i >> 7);
    int c    = (int)(i & (HDIM - 1));
    float d  = dinv[node];
    float v  = d * d * h[i];
    if (bias) v += bias[c];
    out[i] = v;
  }
}

// out[col] += dinv[row]*dinv[col] * h[row] ; one wave per edge, float4/lane.
__global__ void prop_scatter(const int* __restrict__ rows,
                             const int* __restrict__ cols,
                             const float* __restrict__ dinv,
                             const float* __restrict__ h,
                             float* __restrict__ out, int ne) {
  int wid  = blockIdx.x * (blockDim.x >> 5) + (threadIdx.x >> 5);
  int lane = threadIdx.x & 31;
  if (wid >= ne) return;
  int r = rows[wid], c = cols[wid];
  float nrm = dinv[r] * dinv[c];
  const float4 v = ((const float4*)(h + (size_t)r * HDIM))[lane];
  float* op = out + (size_t)c * HDIM + lane * 4;
  atomicAdd(op + 0, nrm * v.x);
  atomicAdd(op + 1, nrm * v.y);
  atomicAdd(op + 2, nrm * v.z);
  atomicAdd(op + 3, nrm * v.w);
}

// h = a*h + b*x0 (in place)
__global__ void axpby_inplace(float* __restrict__ h, const float* __restrict__ x0,
                              float a, float b, size_t total) {
  size_t i = (size_t)blockIdx.x * blockDim.x + threadIdx.x;
  if (i < total) h[i] = a * h[i] + b * x0[i];
}

// --------------------------- B fragment packing ----------------------------
// Source: W[K x 128] fp32 row-major. Dest: bf16 fragments indexed as
//   out[(((t*8 + w)*32 + l)*16 + e)]  =  W[(t*32 + 16*(l>>4) + e)*128 + w*16 + (l&15)]
// so lane l of N-tile w at k-step t holds its 16 B elements contiguously (32B).
__global__ void pack_b_frag(const float* __restrict__ W,
                            __hip_bfloat16* __restrict__ out, int total) {
  int idx = blockIdx.x * blockDim.x + threadIdx.x;   // total = K*128
  if (idx >= total) return;
  int e = idx & 15;
  int l = (idx >> 4) & 31;
  int w = (idx >> 9) & 7;
  int t = idx >> 12;
  int n = (w << 4) + (l & 15);
  int k = (t << 5) + ((l >> 4) << 4) + e;
  out[idx] = __float2bfloat16(W[k * HDIM + n]);
}

// ------------------------------- WMMA GEMM ---------------------------------
// D[N x 128] = accScale * (A[N x K] @ B) + cScale * C[N x 128]
// Block: 256 threads = 8 waves. Block owns 16 rows; wave w owns cols [16w,16w+16).
// Whole 16xK A tile staged once in LDS, bf16, in per-lane fragment order.
template <int K>
__global__ void wmma_gemm_bf16(const float* __restrict__ A,
                               const __hip_bfloat16* __restrict__ Bpk,
                               const float* __restrict__ C,
                               float cScale, float accScale,
                               float* __restrict__ D, int nrows) {
  constexpr int NT = K / 32;                 // k-steps
  __shared__ v16bf AsFrag[NT * 32];          // 32B per lane per k-step
  __bf16* AsB = (__bf16*)AsFrag;

  const int row0 = blockIdx.x << 4;
  const int wave = threadIdx.x >> 5;         // 0..7 -> N tile
  const int lane = threadIdx.x & 31;

  // ---- stage A tile (16 x K fp32 -> bf16 fragments in LDS) ----
  for (int idx = threadIdx.x; idx < 16 * K; idx += 256) {
    int r  = idx / K;                        // row within tile
    int gc = idx & (K - 1);                  // k index (K is pow2)
    int gr = row0 + r;
    float v = (gr < nrows) ? A[(size_t)gr * K + gc] : 0.0f;
    int t    = gc >> 5;
    int cc   = gc & 31;
    int half = cc >> 4;
    int e    = cc & 15;
    int ln   = r + (half << 4);
    AsB[((t << 5) + ln) * 16 + e] = (__bf16)v;
  }
  __syncthreads();

  // ---- main loop: 1 LDS vec load + 1 global vec load per wmma ----
  const v16bf* Bfrag = (const v16bf*)Bpk;
  v8f acc = {};
#pragma unroll
  for (int t = 0; t < NT; ++t) {
    v16bf a = AsFrag[(t << 5) + lane];
    v16bf b = Bfrag[((t << 3) + wave) * 32 + lane];
    acc = __builtin_amdgcn_wmma_f32_16x16x32_bf16(
        /*neg_a=*/false, a, /*neg_b=*/false, b,
        /*c_mod=*/(short)0, acc, /*reuse_a=*/false, /*reuse_b=*/false);
  }

  // ---- epilogue: D element r: M = r + 8*(lane>>4), N = wave*16 + (lane&15) ----
  const int half = lane >> 4;
  const int mn   = lane & 15;
#pragma unroll
  for (int r = 0; r < 8; ++r) {
    int gm = row0 + r + (half << 3);
    if (gm < nrows) {
      size_t off = (size_t)gm * HDIM + (wave << 4) + mn;
      float v = accScale * acc[r];
      if (C) v += cScale * C[off];
      D[off] = v;
    }
  }
}

// -------------------------------- batchnorm --------------------------------

__global__ void bn_reduce(const float* __restrict__ z,
                          float* __restrict__ sums, float* __restrict__ sqs,
                          size_t total) {
  __shared__ float ss[HDIM];
  __shared__ float sq[HDIM];
  int tid = threadIdx.x;
  if (tid < HDIM) { ss[tid] = 0.0f; sq[tid] = 0.0f; }
  __syncthreads();

  size_t i0     = (size_t)blockIdx.x * blockDim.x + tid;
  size_t stride = (size_t)gridDim.x * blockDim.x;   // multiple of 128
  int c = (int)(i0 & (HDIM - 1));                   // fixed column per thread
  float a = 0.0f, b = 0.0f;
  for (size_t i = i0; i < total; i += stride) {
    float v = z[i];
    a += v;
    b += v * v;
  }
  atomicAdd(&ss[c], a);
  atomicAdd(&sq[c], b);
  __syncthreads();
  if (tid < HDIM) {
    atomicAdd(&sums[tid], ss[tid]);
    atomicAdd(&sqs[tid],  sq[tid]);
  }
}

// sums -> scale, sqs -> shift
__global__ void bn_finalize(float* __restrict__ sums, float* __restrict__ sqs,
                            const float* __restrict__ g,
                            const float* __restrict__ b, float ncount) {
  int c = threadIdx.x;
  if (c < HDIM) {
    float mu  = sums[c] / ncount;
    float var = sqs[c] / ncount - mu * mu;
    float sc  = g[c] * rsqrtf(var + EPS_C);
    sums[c] = sc;
    sqs[c]  = b[c] - mu * sc;
  }
}

__global__ void bn_apply(float* __restrict__ z,
                         const float* __restrict__ scale,
                         const float* __restrict__ shift, size_t total) {
  size_t i = (size_t)blockIdx.x * blockDim.x + threadIdx.x;
  if (i < total) {
    int c = (int)(i & (HDIM - 1));
    z[i] = fmaxf(z[i] * scale[c] + shift[c], 0.0f);
  }
}

// --------------------------------- driver ----------------------------------

extern "C" void kernel_launch(void* const* d_in, const int* in_sizes, int n_in,
                              void* d_out, int out_size, void* d_ws, size_t ws_size,
                              hipStream_t stream) {
  const float* x     = (const float*)d_in[0];
  const int*   ei    = (const int*)d_in[1];
  const float* w0    = (const float*)d_in[2];
  const float* b0    = (const float*)d_in[3];
  const float* convw = (const float*)d_in[4];
  const float* gamma = (const float*)d_in[5];
  const float* betap = (const float*)d_in[6];

  const int FIN = 256;
  const int N   = in_sizes[0] / FIN;
  const int E   = in_sizes[1] / 2;
  const int L   = in_sizes[4] / (HDIM * HDIM);
  const size_t NH = (size_t)N * HDIM;

  const int* rows = ei;       // edge_index[0]
  const int* cols = ei + E;   // edge_index[1]

  // --- carve workspace (256B-aligned slices) ---
  char* wsp = (char*)d_ws;
  auto carve = [&](size_t bytes) -> void* {
    void* p = (void*)wsp;
    wsp += (bytes + 255) & ~(size_t)255;
    return p;
  };
  float* T    = (float*)carve(NH * sizeof(float));   // gemm out / h / s
  float* X0   = (float*)carve(NH * sizeof(float));   // initial features z0
  float* dinv = (float*)carve((size_t)N * sizeof(float));
  float* bnS  = (float*)carve(HDIM * sizeof(float));
  float* bnQ  = (float*)carve(HDIM * sizeof(float));
  __hip_bfloat16* Wb0 = (__hip_bfloat16*)carve((size_t)FIN * HDIM * 2);
  __hip_bfloat16* Wbc = (__hip_bfloat16*)carve((size_t)L * HDIM * HDIM * 2);

  float* Z = (float*)d_out;

  const int TB = 256;
  const int wavesPerBlock = TB / 32;

  // 1) weights -> bf16 fragments (once; 200KB total)
  int nw0 = FIN * HDIM;
  pack_b_frag<<<(nw0 + TB - 1) / TB, TB, 0, stream>>>(w0, Wb0, nw0);
  for (int i = 0; i < L; ++i) {
    int nwc = HDIM * HDIM;
    pack_b_frag<<<(nwc + TB - 1) / TB, TB, 0, stream>>>(
        convw + (size_t)i * HDIM * HDIM, Wbc + (size_t)i * HDIM * HDIM, nwc);
  }

  // 2) degrees (self-loop => start at 1), then dinv = rsqrt(deg)
  fill_f32<<<(N + TB - 1) / TB, TB, 0, stream>>>(dinv, 1.0f, (size_t)N);
  deg_accum<<<(E + TB - 1) / TB, TB, 0, stream>>>(cols, dinv, E);
  deg_to_dinv<<<(N + TB - 1) / TB, TB, 0, stream>>>(dinv, N);

  // 3) T = x @ W0   (WMMA bf16, K=256)
  wmma_gemm_bf16<256><<<(N + 15) / 16, TB, 0, stream>>>(
      x, Wb0, nullptr, 0.0f, 1.0f, T, N);

  // 4) X0 = prop(T) + lin0_b   (init with self-loop+bias, scatter edges)
  prop_init<<<(NH + TB - 1) / TB, TB, 0, stream>>>(dinv, T, b0, X0, NH);
  prop_scatter<<<(E + wavesPerBlock - 1) / wavesPerBlock, TB, 0, stream>>>(
      rows, cols, dinv, T, X0, E);

  const float* zcur = X0;
  for (int i = 0; i < L; ++i) {
    // h = prop(zcur) -> T
    prop_init<<<(NH + TB - 1) / TB, TB, 0, stream>>>(dinv, zcur, nullptr, T, NH);
    prop_scatter<<<(E + wavesPerBlock - 1) / wavesPerBlock, TB, 0, stream>>>(
        rows, cols, dinv, zcur, T, E);

    // s = (1-alpha)*h + alpha*x0   (in place in T)
    axpby_inplace<<<(NH + TB - 1) / TB, TB, 0, stream>>>(
        T, X0, 1.0f - ALPHA_C, ALPHA_C, NH);

    // z = beta*(s@Wi) + (1-beta)*s   (fused WMMA epilogue, K=128), into d_out
    float beta = logf(THETA_C / (float)(i + 1) + 1.0f);
    wmma_gemm_bf16<128><<<(N + 15) / 16, TB, 0, stream>>>(
        T, Wbc + (size_t)i * HDIM * HDIM, T, 1.0f - beta, beta, Z, N);

    // batchnorm + relu (all but last layer), in place on Z
    if (i < L - 1) {
      fill_f32<<<1, TB, 0, stream>>>(bnS, 0.0f, HDIM);
      fill_f32<<<1, TB, 0, stream>>>(bnQ, 0.0f, HDIM);
      bn_reduce<<<512, TB, 0, stream>>>(Z, bnS, bnQ, NH);
      bn_finalize<<<1, HDIM, 0, stream>>>(bnS, bnQ, gamma + i * HDIM,
                                          betap + i * HDIM, (float)N);
      bn_apply<<<(NH + TB - 1) / TB, TB, 0, stream>>>(Z, bnS, bnQ, NH);
    }
    zcur = Z;
  }
}
